// VeectorQuantizer_59373627900326
// MI455X (gfx1250) — matrix operand order repaired and checked by reading, compiled, not run on hardware
//
#include <hip/hip_runtime.h>

typedef __attribute__((ext_vector_type(16))) __bf16          v16bf;
typedef __attribute__((ext_vector_type(16))) unsigned short  v16u;
typedef __attribute__((ext_vector_type(8)))  float           v8f;

#define HD     256
#define KCODES 1024
#define NROWS  32768

__device__ __forceinline__ unsigned short f2bf_rne(float f) {
  unsigned int u = __builtin_bit_cast(unsigned int, f);
  u += 0x7FFFu + ((u >> 16) & 1u);
  return (unsigned short)(u >> 16);
}
__device__ __forceinline__ float bf2f(unsigned short h) {
  unsigned int u = ((unsigned int)h) << 16;
  return __builtin_bit_cast(float, u);
}

// --- Kernel 1: split codebook into bf16 hi/lo planes, row norms, zero acc ---
__global__ void vq_prep(const float* __restrict__ emb,
                        unsigned short* __restrict__ ehi,
                        unsigned short* __restrict__ elo,
                        float* __restrict__ enorm,
                        float* __restrict__ acc) {
  __shared__ float red[HD];
  const int n = blockIdx.x, t = threadIdx.x;
  float v = emb[n * HD + t];
  unsigned short h = f2bf_rne(v);
  ehi[n * HD + t] = h;
  elo[n * HD + t] = f2bf_rne(v - bf2f(h));
  red[t] = v * v;
  __syncthreads();
  for (int s = HD / 2; s > 0; s >>= 1) {
    if (t < s) red[t] += red[t + s];
    __syncthreads();
  }
  if (t == 0) enorm[n] = red[0];
  if (n == 0 && t == 0) *acc = 0.0f;
}

// --- Kernel 2: distances via bf16-split WMMA, argmin, gather, loss partial ---
__global__ void __launch_bounds__(128) vq_main(
    const float* __restrict__ z, const float* __restrict__ emb,
    const unsigned short* __restrict__ ehi, const unsigned short* __restrict__ elo,
    const float* __restrict__ enorm, float* __restrict__ out,
    float* __restrict__ acc) {
  __shared__ int   s_idx[4][16];
  __shared__ float s_red[128];
  const int tid  = threadIdx.x;
  const int warp = tid >> 5;
  const int lane = tid & 31;
  const int m    = lane & 15;   // A row / B column / C column within tile
  const int half = lane >> 4;
  const int row0 = (blockIdx.x * 4 + warp) * 16;

  // A fragments: 16 z rows x 256, split into bf16 hi/lo, held in VGPRs.
  // 16-bit A layout: lane m, elems 0..7 <- K = kb*32 + 16*half + j,
  //                          elems 8..15 <- K = kb*32 + 16*half + 8 + j.
  v16u ahi[8], alo[8];
  const float* zr = z + (size_t)(row0 + m) * HD;
#pragma unroll
  for (int kb = 0; kb < 8; ++kb) {
#pragma unroll
    for (int j = 0; j < 8; ++j) {
      float v0 = zr[kb * 32 + half * 16 + j];
      unsigned short h0 = f2bf_rne(v0);
      ahi[kb][j] = h0;
      alo[kb][j] = f2bf_rne(v0 - bf2f(h0));
      float v1 = zr[kb * 32 + half * 16 + 8 + j];
      unsigned short h1 = f2bf_rne(v1);
      ahi[kb][j + 8] = h1;
      alo[kb][j + 8] = f2bf_rne(v1 - bf2f(h1));
    }
  }

  float bestd[8];
  int   besti[8];
#pragma unroll
  for (int r = 0; r < 8; ++r) { bestd[r] = 3.4e38f; besti[r] = 0; }

  for (int nt = 0; nt < KCODES / 16; ++nt) {
    const int ng = nt * 16 + m;  // this lane's codebook column
    // B layout (32x16): lanes 0-15 hold K=0..15, lanes 16-31 K=16..31.
    const unsigned short* bh = ehi + (size_t)ng * HD + half * 16;
    const unsigned short* bl = elo + (size_t)ng * HD + half * 16;
    v8f c = {};
#pragma unroll
    for (int kb = 0; kb < 8; ++kb) {
      v16bf bhv = *reinterpret_cast<const v16bf*>(bh + kb * 32);
      v16bf blv = *reinterpret_cast<const v16bf*>(bl + kb * 32);
      v16bf ah = __builtin_bit_cast(v16bf, ahi[kb]);
      v16bf al = __builtin_bit_cast(v16bf, alo[kb]);
      c = __builtin_amdgcn_wmma_f32_16x16x32_bf16(false, ah, false, bhv,
                                                  (short)0, c, false, false);
      c = __builtin_amdgcn_wmma_f32_16x16x32_bf16(false, ah, false, blv,
                                                  (short)0, c, false, false);
      c = __builtin_amdgcn_wmma_f32_16x16x32_bf16(false, al, false, bhv,
                                                  (short)0, c, false, false);
    }
    const float en = enorm[ng];
#pragma unroll
    for (int r = 0; r < 8; ++r) {
      // ||z||^2 is constant per row: argmin over (||e||^2 - 2 z.e)
      float s = __builtin_fmaf(-2.0f, c[r], en);
      if (s < bestd[r]) { bestd[r] = s; besti[r] = ng; }  // keeps lowest ng on tie
    }
  }

  // Argmin across the 16 lanes of this half (C row r maps to z row r + 8*half).
#pragma unroll
  for (int r = 0; r < 8; ++r) {
    float d = bestd[r];
    int   i = besti[r];
#pragma unroll
    for (int off = 1; off < 16; off <<= 1) {
      float od = __shfl_xor(d, off, 32);
      int   oi = __shfl_xor(i, off, 32);
      if (od < d || (od == d && oi < i)) { d = od; i = oi; }
    }
    bestd[r] = d;
    besti[r] = i;
  }
  if (m == 0) {
#pragma unroll
    for (int r = 0; r < 8; ++r) s_idx[warp][half * 8 + r] = besti[r];
  }
  __syncthreads();

  // Gather z_q = embedding[idx], write out, accumulate squared error.
  const int rowb = blockIdx.x * 64;
  const int c0   = tid * 2;
  float lsum = 0.0f;
  for (int rl = 0; rl < 64; ++rl) {
    const int idx = s_idx[rl >> 4][rl & 15];
    const float2 ev = *reinterpret_cast<const float2*>(emb + (size_t)idx * HD + c0);
    const float2 zv = *reinterpret_cast<const float2*>(z + (size_t)(rowb + rl) * HD + c0);
    *reinterpret_cast<float2*>(out + (size_t)(rowb + rl) * HD + c0) = ev;
    const float dx = ev.x - zv.x, dy = ev.y - zv.y;
    lsum = __builtin_fmaf(dx, dx, lsum);
    lsum = __builtin_fmaf(dy, dy, lsum);
  }
  s_red[tid] = lsum;
  __syncthreads();
  for (int s = 64; s > 0; s >>= 1) {
    if (tid < s) s_red[tid] += s_red[tid + s];
    __syncthreads();
  }
  if (tid == 0) atomicAdd(acc, s_red[0]);
}

// --- Kernel 3: finalize loss = 1.25 * mean ---
__global__ void vq_final(const float* __restrict__ acc, float* __restrict__ loss) {
  *loss = 1.25f * (*acc) / (float)(NROWS * HD);
}

extern "C" void kernel_launch(void* const* d_in, const int* in_sizes, int n_in,
                              void* d_out, int out_size, void* d_ws, size_t ws_size,
                              hipStream_t stream) {
  const float* z   = (const float*)d_in[0];  // [32,32,32,256] fp32
  const float* emb = (const float*)d_in[1];  // [1024,256] fp32
  float* out = (float*)d_out;                // z_q flat (32768*256) + loss (1)

  char* ws = (char*)d_ws;
  unsigned short* ehi   = (unsigned short*)ws;                        // 512 KB
  unsigned short* elo   = (unsigned short*)(ws + 512 * 1024);         // 512 KB
  float*          enorm = (float*)(ws + 1024 * 1024);                 // 4 KB
  float*          acc   = (float*)(ws + 1024 * 1024 + 4096);          // 4 B

  vq_prep<<<KCODES, HD, 0, stream>>>(emb, ehi, elo, enorm, acc);
  vq_main<<<NROWS / 64, 128, 0, stream>>>(z, emb, ehi, elo, enorm, out, acc);
  vq_final<<<1, 1, 0, stream>>>(acc, out + (size_t)NROWS * HD);
}
